// TENFFTLayer_7017976561862
// MI455X (gfx1250) — matrix operand order repaired and checked by reading, not compiled, not run on hardware
//
#include <hip/hip_runtime.h>
#include <hip/hip_bf16.h>
#include <stdint.h>

// ---------------------------------------------------------------- constants
#define BB   32
#define TT   1024
#define DD   512
#define KK   64
#define HH   4
#define KP   16
#define MLPD 2048
#define BT   (BB * TT)          // 32768 rows

// GEMM tiling: block tile 256(M) x 128(N), BK=32, 8 waves in 4(M)x2(N) grid,
// wave tile 64x64 = 4x4 accumulators of 16x16.
#define BM 256
#define BN 128
#define BK 32
#define LDSP 40                 // padded LDS row stride (bf16 elems), 80B rows

typedef __attribute__((ext_vector_type(16))) __bf16 v16bf;
typedef __attribute__((ext_vector_type(8)))  float  v8f;

__device__ __forceinline__ unsigned short f32_to_bf16(float f) {
    unsigned int u = __float_as_uint(f);
    unsigned int r = (u + 0x7FFFu + ((u >> 16) & 1u)) >> 16;
    return (unsigned short)r;
}

__device__ __forceinline__ float sigmoidf_dev(float x) {
    return 1.0f / (1.0f + __expf(-x));
}

// CDNA5 async DMA: global (16B per lane) -> LDS, tracked by ASYNCcnt.
__device__ __forceinline__ void async_copy16(const void* gptr, unsigned lds_addr) {
    asm volatile("global_load_async_to_lds_b128 %0, %1, off"
                 :: "v"(lds_addr), "v"(gptr)
                 : "memory");
}

// ---------------------------------------------------------------- weight transpose + f32->bf16
// src: [R, C] f32 row-major  ->  dst: [C, R] bf16 row-major
__global__ void k_transpose_bf16(const float* __restrict__ src,
                                 unsigned short* __restrict__ dst,
                                 int R, int C) {
    int i = blockIdx.x * blockDim.x + threadIdx.x;
    if (i >= R * C) return;
    int r = i / C, c = i % C;
    dst[(size_t)c * R + r] = f32_to_bf16(src[i]);
}

// ---------------------------------------------------------------- LayerNorm -> bf16 (row per block, 128 thr)
__global__ __launch_bounds__(128) void k_layernorm_bf16(
    const float* __restrict__ x, const float* __restrict__ g,
    const float* __restrict__ b, unsigned short* __restrict__ out) {
    __shared__ float red[128];
    const int row = blockIdx.x, t = threadIdx.x;
    const float* xr = x + (size_t)row * DD;
    float v[4]; float s = 0.f;
    #pragma unroll
    for (int i = 0; i < 4; ++i) { v[i] = xr[t + i * 128]; s += v[i]; }
    red[t] = s; __syncthreads();
    for (int off = 64; off > 0; off >>= 1) { if (t < off) red[t] += red[t + off]; __syncthreads(); }
    float mean = red[0] * (1.0f / DD); __syncthreads();
    float sq = 0.f;
    #pragma unroll
    for (int i = 0; i < 4; ++i) { float d = v[i] - mean; sq += d * d; }
    red[t] = sq; __syncthreads();
    for (int off = 64; off > 0; off >>= 1) { if (t < off) red[t] += red[t + off]; __syncthreads(); }
    float inv = rsqrtf(red[0] * (1.0f / DD) + 1e-5f);
    #pragma unroll
    for (int i = 0; i < 4; ++i) {
        int n = t + i * 128;
        out[(size_t)row * DD + n] = f32_to_bf16((v[i] - mean) * inv * g[n] + b[n]);
    }
}

// ---------------------------------------------------------------- complex first-order recurrence
__global__ __launch_bounds__(256) void k_scan(
    const float* __restrict__ beta, const float* __restrict__ log_decay,
    const float* __restrict__ freq, float* __restrict__ cbuf) {
    int tid = blockIdx.x * blockDim.x + threadIdx.x;   // 0 .. B*K-1
    if (tid >= BB * KK) return;
    int b = tid / KK, k = tid % KK;
    float mag = sigmoidf_dev(log_decay[k]);
    float lr = mag * __cosf(freq[k]);
    float li = mag * __sinf(freq[k]);
    float cr = 0.f, ci = 0.f;
    size_t base = (size_t)b * TT * (2 * KK);
    for (int t = 0; t < TT; ++t) {
        size_t o = base + (size_t)t * (2 * KK);
        float br = beta[o + k], bi = beta[o + KK + k];
        float nr = lr * cr - li * ci + br;
        float ni = lr * ci + li * cr + bi;
        cr = nr; ci = ni;
        cbuf[o + k] = cr; cbuf[o + KK + k] = ci;
    }
}

// ---------------------------------------------------------------- per-head coupling (16x16 per head)
__global__ __launch_bounds__(256) void k_coupling(
    const float* __restrict__ cbuf, const float* __restrict__ coupling,
    float* __restrict__ eig_f32, unsigned short* __restrict__ eig_bf) {
    int tid = blockIdx.x * blockDim.x + threadIdx.x;
    if (tid >= BT * 2 * KK) return;
    int col = tid & (2 * KK - 1);
    int bt  = tid >> 7;                 // 2K = 128
    int ri  = col >> 6;                 // 0 real / 1 imag
    int w   = col & 63;
    int h = w >> 4, j = w & 15;
    const float* cp = coupling + (size_t)h * KP * KP + (size_t)j * KP;
    const float* cv = cbuf + (size_t)bt * (2 * KK) + ri * KK + h * KP;
    float s = 0.f;
    #pragma unroll
    for (int k = 0; k < KP; ++k) s += cp[k] * cv[k];
    eig_f32[(size_t)bt * (2 * KK) + col] = s;
    eig_bf [(size_t)bt * (2 * KK) + col] = f32_to_bf16(s);
}

// ---------------------------------------------------------------- gate + residual + LN2
__global__ __launch_bounds__(128) void k_gate_x1_ln(
    const float* __restrict__ x, const float* __restrict__ gp,
    const float* __restrict__ h, const float* __restrict__ b_gate,
    const float* __restrict__ g2, const float* __restrict__ bn2,
    float* __restrict__ x1, unsigned short* __restrict__ xn2) {
    __shared__ float red[128];
    const int row = blockIdx.x, t = threadIdx.x;
    size_t base = (size_t)row * DD;
    float v[4]; float s = 0.f;
    #pragma unroll
    for (int i = 0; i < 4; ++i) {
        int n = t + i * 128;
        float gate = sigmoidf_dev(gp[base + n] + b_gate[n]);
        v[i] = x[base + n] + gate * h[base + n];
        x1[base + n] = v[i];
        s += v[i];
    }
    red[t] = s; __syncthreads();
    for (int off = 64; off > 0; off >>= 1) { if (t < off) red[t] += red[t + off]; __syncthreads(); }
    float mean = red[0] * (1.0f / DD); __syncthreads();
    float sq = 0.f;
    #pragma unroll
    for (int i = 0; i < 4; ++i) { float d = v[i] - mean; sq += d * d; }
    red[t] = sq; __syncthreads();
    for (int off = 64; off > 0; off >>= 1) { if (t < off) red[t] += red[t + off]; __syncthreads(); }
    float inv = rsqrtf(red[0] * (1.0f / DD) + 1e-5f);
    #pragma unroll
    for (int i = 0; i < 4; ++i) {
        int n = t + i * 128;
        xn2[base + n] = f32_to_bf16((v[i] - mean) * inv * g2[n] + bn2[n]);
    }
}

// ---------------------------------------------------------------- WMMA GEMM with async global->LDS double buffering
// C[M,N] = A[M,Kd] * Bt[N,Kd]^T, A/Bt bf16, acc f32.
// mode 0: Cf = acc
// mode 1: Cb = bf16(silu(acc + bias[n]))
// mode 2: Cf = add + acc + bias[n]
__global__ __launch_bounds__(256) void k_gemm_wmma(
    const unsigned short* __restrict__ A, const unsigned short* __restrict__ Bt,
    int M, int N, int Kd, int mode,
    float* __restrict__ Cf, unsigned short* __restrict__ Cb,
    const float* __restrict__ bias, const float* __restrict__ add) {
    __shared__ __align__(16) unsigned short As[2][BM * LDSP];   // 2 x 256x32 bf16 (padded)
    __shared__ __align__(16) unsigned short Bs[2][BN * LDSP];   // 2 x 128x32 bf16 (padded)

    const int tid  = threadIdx.x;
    const int lane = tid & 31;
    const int wave = tid >> 5;
    const int waveM = wave & 3;         // 0..3  (64 rows each)
    const int waveN = wave >> 2;        // 0..1  (64 cols each)
    const int tileM = blockIdx.y * BM;
    const int tileN = blockIdx.x * BN;
    const int hl = lane >> 4;           // half-wave select
    const int lr = lane & 15;

    // per-thread async copy assignments (6 x 16B per stage)
    const int ra = tid >> 2;            // 0..63  -> rows ra, ra+64, ra+128, ra+192 of A tile
    const int ca = (tid & 3) * 8;       // bf16 col chunk
    const int rb = tid >> 1;            // 0..127 -> row of B tile
    const int cb = (tid & 1) * 16;      // bf16 col base (two 8-elem chunks)

    const int S = Kd / BK;

    auto issue_stage = [&](int s, int buf) {
        const int k0 = s * BK;
        #pragma unroll
        for (int p = 0; p < 4; ++p) {
            int row = ra + p * 64;
            async_copy16(A + (size_t)(tileM + row) * Kd + k0 + ca,
                         (unsigned)(uintptr_t)(&As[buf][row * LDSP + ca]));
        }
        #pragma unroll
        for (int p = 0; p < 2; ++p) {
            async_copy16(Bt + (size_t)(tileN + rb) * Kd + k0 + cb + p * 8,
                         (unsigned)(uintptr_t)(&Bs[buf][rb * LDSP + cb + p * 8]));
        }
    };

    v8f acc[4][4] = {};

    issue_stage(0, 0);

    for (int s = 0; s < S; ++s) {
        const int cur = s & 1;
        if (s + 1 < S) {
            issue_stage(s + 1, cur ^ 1);
            asm volatile("s_wait_asynccnt 0x6" ::: "memory");   // stage s copies done
        } else {
            asm volatile("s_wait_asynccnt 0x0" ::: "memory");
        }
        __syncthreads();

        // fragments per ISA layout:
        // A (16x32): lane lr = row M; halves own K {hl*8..+7, 16+hl*8..+7}
        // B (32x16): lane lr = col N; halves own K {hl*8..+7, 16+hl*8..+7}
        union Frag { v16bf v; uint4 q[2]; } a[4];
        #pragma unroll
        for (int i = 0; i < 4; ++i) {
            int row = waveM * 64 + i * 16 + lr;
            a[i].q[0] = *(const uint4*)(&As[cur][row * LDSP + hl * 8]);
            a[i].q[1] = *(const uint4*)(&As[cur][row * LDSP + 16 + hl * 8]);
        }
        #pragma unroll
        for (int j = 0; j < 4; ++j) {
            union Frag b;
            int row = waveN * 64 + j * 16 + lr;
            b.q[0] = *(const uint4*)(&Bs[cur][row * LDSP + hl * 8]);
            b.q[1] = *(const uint4*)(&Bs[cur][row * LDSP + 16 + hl * 8]);
            #pragma unroll
            for (int i = 0; i < 4; ++i)
                acc[i][j] = __builtin_amdgcn_wmma_f32_16x16x32_bf16(
                    false, a[i].v, false, b.v, (short)0, acc[i][j], false, false);
        }
        __syncthreads();   // all reads of buffer `cur` retired before it is overwritten
    }

    // epilogue; C layout: VGPR g -> M = g + 8*hl, N = lr
    #pragma unroll
    for (int i = 0; i < 4; ++i) {
        #pragma unroll
        for (int j = 0; j < 4; ++j) {
            int baseM = tileM + waveM * 64 + i * 16 + hl * 8;
            int n     = tileN + waveN * 64 + j * 16 + lr;
            #pragma unroll
            for (int g = 0; g < 8; ++g) {
                size_t idx = (size_t)(baseM + g) * N + n;
                float v = acc[i][j][g];
                if (mode == 0) {
                    Cf[idx] = v;
                } else if (mode == 1) {
                    float t = v + bias[n];
                    Cb[idx] = f32_to_bf16(t * sigmoidf_dev(t));
                } else {
                    Cf[idx] = add[idx] + v + bias[n];
                }
            }
        }
    }
}

// ---------------------------------------------------------------- host launch
extern "C" void kernel_launch(void* const* d_in, const int* in_sizes, int n_in,
                              void* d_out, int out_size, void* d_ws, size_t ws_size,
                              hipStream_t stream) {
    const float* x         = (const float*)d_in[0];   // [B,T,D]
    const float* w_in      = (const float*)d_in[1];   // [D,2K]
    const float* log_decay = (const float*)d_in[2];   // [K]
    const float* frequency = (const float*)d_in[3];   // [K]
    const float* coupling  = (const float*)d_in[4];   // [H,KP,KP]
    const float* w_out     = (const float*)d_in[5];   // [2K,D]
    const float* w_gate    = (const float*)d_in[6];   // [D,D]
    const float* b_gate    = (const float*)d_in[7];   // [D]
    const float* w1        = (const float*)d_in[8];   // [D,MLP]
    const float* b1        = (const float*)d_in[9];   // [MLP]
    const float* w2        = (const float*)d_in[10];  // [MLP,D]
    const float* b2        = (const float*)d_in[11];  // [D]
    const float* g1        = (const float*)d_in[12];
    const float* bn1       = (const float*)d_in[13];
    const float* g2        = (const float*)d_in[14];
    const float* bn2       = (const float*)d_in[15];

    float* out_main = (float*)d_out;                        // [BT, D]
    float* out_eig  = (float*)d_out + (size_t)BT * DD;      // [BT, 2K]

    // workspace carving (256B aligned)
    char* ws = (char*)d_ws;
    size_t off = 0;
    auto carve = [&](size_t bytes) -> char* {
        char* p = ws + off;
        off = (off + bytes + 255) & ~(size_t)255;
        return p;
    };
    unsigned short* wInT   = (unsigned short*)carve((size_t)(2*KK) * DD * 2); // [2K, D]
    unsigned short* wGateT = (unsigned short*)carve((size_t)DD * DD * 2);     // [D, D]
    unsigned short* wOutT  = (unsigned short*)carve((size_t)DD * (2*KK) * 2); // [D, 2K]
    unsigned short* w1T    = (unsigned short*)carve((size_t)MLPD * DD * 2);   // [MLP, D]
    unsigned short* w2T    = (unsigned short*)carve((size_t)DD * MLPD * 2);   // [D, MLP]
    unsigned short* xn_bf  = (unsigned short*)carve((size_t)BT * DD * 2);
    float*          beta   = (float*)carve((size_t)BT * (2*KK) * 4);
    float*          cbuf   = (float*)carve((size_t)BT * (2*KK) * 4);
    unsigned short* eig_bf = (unsigned short*)carve((size_t)BT * (2*KK) * 2);
    float*          hbuf   = (float*)carve((size_t)BT * DD * 4);
    float*          gpbuf  = (float*)carve((size_t)BT * DD * 4);
    float*          x1buf  = (float*)carve((size_t)BT * DD * 4);
    unsigned short* xn2_bf = (unsigned short*)carve((size_t)BT * DD * 2);
    unsigned short* mlp_bf = (unsigned short*)carve((size_t)BT * MLPD * 2);
    (void)ws_size; (void)in_sizes; (void)n_in; (void)out_size;

    // 1) weights -> transposed bf16
    auto tgrid = [](int e) { return (e + 255) / 256; };
    hipLaunchKernelGGL(k_transpose_bf16, dim3(tgrid(DD * 2 * KK)), dim3(256), 0, stream, w_in,   wInT,   DD,   2*KK);
    hipLaunchKernelGGL(k_transpose_bf16, dim3(tgrid(DD * DD)),     dim3(256), 0, stream, w_gate, wGateT, DD,   DD);
    hipLaunchKernelGGL(k_transpose_bf16, dim3(tgrid(2*KK * DD)),   dim3(256), 0, stream, w_out,  wOutT,  2*KK, DD);
    hipLaunchKernelGGL(k_transpose_bf16, dim3(tgrid(DD * MLPD)),   dim3(256), 0, stream, w1,     w1T,    DD,   MLPD);
    hipLaunchKernelGGL(k_transpose_bf16, dim3(tgrid(MLPD * DD)),   dim3(256), 0, stream, w2,     w2T,    MLPD, DD);

    // 2) LN1 -> xn bf16
    hipLaunchKernelGGL(k_layernorm_bf16, dim3(BT), dim3(128), 0, stream, x, g1, bn1, xn_bf);

    // 3) beta = xn @ w_in          [BT,512] x [512,128]
    hipLaunchKernelGGL(k_gemm_wmma, dim3((2*KK)/BN, BT/BM), dim3(256), 0, stream,
                       xn_bf, wInT, BT, 2*KK, DD, 0, beta, (unsigned short*)nullptr,
                       (const float*)nullptr, (const float*)nullptr);

    // 4) complex recurrence scan
    hipLaunchKernelGGL(k_scan, dim3((BB*KK + 255)/256), dim3(256), 0, stream,
                       beta, log_decay, frequency, cbuf);

    // 5) per-head coupling -> eigenstates (f32 to d_out + bf16 for GEMM)
    hipLaunchKernelGGL(k_coupling, dim3((BT*2*KK)/256), dim3(256), 0, stream,
                       cbuf, coupling, out_eig, eig_bf);

    // 6) h = eig @ w_out           [BT,128] x [128,512]
    hipLaunchKernelGGL(k_gemm_wmma, dim3(DD/BN, BT/BM), dim3(256), 0, stream,
                       eig_bf, wOutT, BT, DD, 2*KK, 0, hbuf, (unsigned short*)nullptr,
                       (const float*)nullptr, (const float*)nullptr);

    // 7) gate_pre = xn @ w_gate    [BT,512] x [512,512]
    hipLaunchKernelGGL(k_gemm_wmma, dim3(DD/BN, BT/BM), dim3(256), 0, stream,
                       xn_bf, wGateT, BT, DD, DD, 0, gpbuf, (unsigned short*)nullptr,
                       (const float*)nullptr, (const float*)nullptr);

    // 8) x1 = x + sigmoid(gate_pre + b_gate)*h ; LN2 -> xn2 bf16
    hipLaunchKernelGGL(k_gate_x1_ln, dim3(BT), dim3(128), 0, stream,
                       x, gpbuf, hbuf, b_gate, g2, bn2, x1buf, xn2_bf);

    // 9) mlp1 = silu(xn2 @ w1 + b1) -> bf16   [BT,512] x [512,2048]
    hipLaunchKernelGGL(k_gemm_wmma, dim3(MLPD/BN, BT/BM), dim3(256), 0, stream,
                       xn2_bf, w1T, BT, MLPD, DD, 1, (float*)nullptr, mlp_bf,
                       b1, (const float*)nullptr);

    // 10) out = x1 + mlp1 @ w2 + b2           [BT,2048] x [2048,512]
    hipLaunchKernelGGL(k_gemm_wmma, dim3(DD/BN, BT/BM), dim3(256), 0, stream,
                       mlp_bf, w2T, BT, DD, MLPD, 2, out_main, (unsigned short*)nullptr,
                       b2, x1buf);
}
